// CrossGraphNodeAttention_15126874816759
// MI455X (gfx1250) — matrix operand order repaired and checked by reading, not compiled, not run on hardware
//
#include <hip/hip_runtime.h>

typedef __attribute__((ext_vector_type(16))) __bf16 v16bf;
typedef __attribute__((ext_vector_type(8)))  float  v8f;

#define HID    256
#define NBATCH 8
#define NA     2048
#define NBK    2048
#define MROWS  (NBATCH * NA)    // 16384 Q rows total
#define KROWS  (NBATCH * NBK)   // 16384 K/V rows total

static __device__ __forceinline__ v8f wmma_bf16(v16bf a, v16bf b, v8f c) {
    // D = A(16x32 bf16) x B(32x16 bf16) + C(16x16 f32)
    return __builtin_amdgcn_wmma_f32_16x16x32_bf16(
        /*neg_a=*/false, a, /*neg_b=*/false, b,
        /*c_mod=*/(short)0, c, /*reuse_a=*/false, /*reuse_b=*/false);
}

// Async global -> LDS copy of 16 bytes (CDNA5 GLOBAL_LOAD_ASYNC_TO_LDS_B128,
// tracked by ASYNCcnt).  vdst VGPR carries the LDS byte address (GV mode).
static __device__ __forceinline__ void async_copy_b128(unsigned lds_addr, const void* gptr) {
    unsigned long long ga = (unsigned long long)(size_t)gptr;
    asm volatile("global_load_async_to_lds_b128 %0, %1, off"
                 :: "v"(lds_addr), "v"(ga) : "memory");
}
#define S_WAIT_ASYNCCNT(n) asm volatile("s_wait_asynccnt %0" :: "i"(n) : "memory")

// low 32 bits of a generic pointer to __shared__ == LDS byte offset
static __device__ __forceinline__ unsigned lds_offset(const void* p) {
    return (unsigned)(size_t)p;
}

// ---------------------------------------------------------------------------
// Kernel 1: QKV projections.  out = X @ W^T + bias, stored bf16.
// One wave computes one 16x16 output tile; K=256 -> 8 WMMA(16x16x32) ops.
// proj 0: Q row-major, proj 1: K row-major, proj 2: V stored transposed
// [b][h][n] so that the attention P*V B-fragments are contiguous loads.
// ---------------------------------------------------------------------------
__global__ __launch_bounds__(256) void qkv_proj(
    const float* __restrict__ A,  const float* __restrict__ Bx,
    const float* __restrict__ Wq, const float* __restrict__ bq,
    const float* __restrict__ Wk, const float* __restrict__ bk,
    const float* __restrict__ Wv, const float* __restrict__ bv,
    __bf16* __restrict__ Qo, __bf16* __restrict__ Ko, __bf16* __restrict__ Vt)
{
    const int lane = threadIdx.x & 31;
    const int wid  = threadIdx.x >> 5;
    const int gw   = blockIdx.x * 8 + wid;

    const int tilesPerProj = (MROWS / 16) * (HID / 16);   // 16384
    const int proj = gw / tilesPerProj;
    const int rem  = gw % tilesPerProj;
    const int m0 = (rem / (HID / 16)) * 16;
    const int n0 = (rem % (HID / 16)) * 16;
    const int cl = lane & 15, half = lane >> 4;

    const float* X    = (proj == 0) ? A  : Bx;
    const float* W    = (proj == 0) ? Wq : (proj == 1) ? Wk : Wv;
    const float* bias = (proj == 0) ? bq : (proj == 1) ? bk : bv;

    v8f acc = {};
    const int rowA = m0 + cl;   // input row
    const int rowB = n0 + cl;   // W row == output column
#pragma unroll
    for (int kc = 0; kc < 8; ++kc) {
        const int k0 = kc * 32;
        v16bf af, bf;
        // A fragment (16x32, 16-bit): lane half selects K-offset 16;
        // slots 0..15 -> K = k0+16h+0..15
        const float* pa = X + (size_t)rowA * HID + k0 + 16 * half;
#pragma unroll
        for (int i = 0; i < 16; ++i) { af[i] = (__bf16)pa[i]; }
        // B fragment (32x16): B[k][n] = W[n][k]; lane half selects K-offset 16,
        // slots contiguous -> 16 contiguous floats of W row n.
        const float* pb = W + (size_t)rowB * HID + k0 + 16 * half;
#pragma unroll
        for (int i = 0; i < 16; ++i) { bf[i] = (__bf16)pb[i]; }
        acc = wmma_bf16(af, bf, acc);
    }

    const int   col  = n0 + cl;
    const float bcol = bias[col];
#pragma unroll
    for (int r = 0; r < 8; ++r) {
        const int rowg = m0 + r + 8 * half;          // C layout: row r (+8 for upper half)
        const __bf16 v = (__bf16)(acc[r] + bcol);
        if (proj == 0)      Qo[(size_t)rowg * HID + col] = v;
        else if (proj == 1) Ko[(size_t)rowg * HID + col] = v;
        else {
            const int bb = rowg / NBK, n = rowg % NBK;
            Vt[((size_t)bb * HID + col) * NBK + n] = v;   // transposed store
        }
    }
}

// ---------------------------------------------------------------------------
// Kernel 2: flash attention.  8 waves/block share one batch; K/V tiles are
// staged into LDS with CDNA5 async global->LDS DMA, double-buffered so the
// copy of tile j+1 overlaps WMMA work on tile j.  Each wave owns 16 Q rows
// (registers), online softmax in f32, P goes through a per-wave LDS buffer
// to convert C-layout -> A-fragment layout.
// ---------------------------------------------------------------------------
__global__ __launch_bounds__(256) void flash_attn(
    const __bf16* __restrict__ Q, const __bf16* __restrict__ K,
    const __bf16* __restrict__ Vt, const int* __restrict__ mask,
    float* __restrict__ out)
{
    __shared__ __align__(16) __bf16 ldsK[2][32 * HID];   // 2 x 16 KB  K tile [32][256]
    __shared__ __align__(16) __bf16 ldsV[2][HID * 32];   // 2 x 16 KB  V^T tile [256][32]
    __shared__ __align__(16) __bf16 ldsP[8][16 * 32];    // 8 KB, per-wave P staging

    const int tid  = threadIdx.x;
    const int lane = tid & 31;
    const int wid  = tid >> 5;
    const int gw   = blockIdx.x * 8 + wid;
    const int b    = gw / (NA / 16);                     // uniform across the block
    const int m0   = (gw % (NA / 16)) * 16;
    const int cl   = lane & 15, half = lane >> 4;

    const __bf16* Kbase = K  + (size_t)b * NBK * HID;
    const __bf16* Vbase = Vt + (size_t)b * HID * NBK;
    const int*    mrow  = mask + b * NBK;

    // Q tile 16x256 resident in registers as 8 A-fragments (64 VGPRs)
    v16bf qf[8];
    {
        const __bf16* qrow = Q + ((size_t)(b * NA + m0 + cl)) * HID;
#pragma unroll
        for (int kc = 0; kc < 8; ++kc) {
            const __bf16* p = qrow + kc * 32 + 16 * half;
#pragma unroll
            for (int i = 0; i < 16; ++i) { qf[kc][i] = p[i]; }
        }
    }

    const v8f vzero = {};
    v8f o[16];
#pragma unroll
    for (int t = 0; t < 16; ++t) o[t] = vzero;

    float m_r[8], l_r[8];
#pragma unroll
    for (int r = 0; r < 8; ++r) { m_r[r] = -3.0e38f; l_r[r] = 0.0f; }

    const unsigned lK0 = lds_offset(&ldsK[0][0]) + (unsigned)tid * 64u;
    const unsigned lK1 = lds_offset(&ldsK[1][0]) + (unsigned)tid * 64u;
    const unsigned lV0 = lds_offset(&ldsV[0][0]) + (unsigned)tid * 64u;
    const unsigned lV1 = lds_offset(&ldsV[1][0]) + (unsigned)tid * 64u;

    // issue async staging of one 32-key tile: K rows are a contiguous 16 KB
    // chunk; V^T contributes 64 contiguous bytes per h-row (h = tid).
    auto stage_tile = [&](int j0, int buf) {
        const char* gk = (const char*)(Kbase + (size_t)j0 * HID) + (size_t)tid * 64;
        const char* gv = (const char*)(Vbase + (size_t)tid * NBK + j0);
        const unsigned lk = buf ? lK1 : lK0;
        const unsigned lv = buf ? lV1 : lV0;
#pragma unroll
        for (int i = 0; i < 4; ++i) async_copy_b128(lk + 16u * i, gk + 16 * i);
#pragma unroll
        for (int i = 0; i < 4; ++i) async_copy_b128(lv + 16u * i, gv + 16 * i);
    };

    const float scl = 0.0625f;   // 1/sqrt(256)
    const int   NT  = NBK / 32;  // 64 tiles

    stage_tile(0, 0);            // prologue: tile 0 in flight

    for (int jt = 0; jt < NT; ++jt) {
        const int j0  = jt * 32;
        const int cur = jt & 1;
        if (jt + 1 < NT) {                 // prefetch next tile, then wait for
            stage_tile(j0 + 32, cur ^ 1);  // current (async loads finish in order)
            S_WAIT_ASYNCCNT(8);
        } else {
            S_WAIT_ASYNCCNT(0);
        }
        __syncthreads();                   // tile `cur` visible to all waves

        const __bf16* Kt = &ldsK[cur][0];  // [32][256]
        const __bf16* Vl = &ldsV[cur][0];  // [256][32]

        // ---- scores: S[16x32] = Q(16x256) . K_tile(32x256)^T (from LDS) ----
        v8f s0 = vzero, s1 = vzero;
#pragma unroll
        for (int kc = 0; kc < 8; ++kc) {
            v16bf kb0, kb1;
            const __bf16* p0 = Kt + (size_t)cl        * HID + kc * 32 + 16 * half;
            const __bf16* p1 = Kt + (size_t)(16 + cl) * HID + kc * 32 + 16 * half;
#pragma unroll
            for (int i = 0; i < 16; ++i) { kb0[i] = p0[i]; kb1[i] = p1[i]; }
            s0 = wmma_bf16(qf[kc], kb0, s0);
            s1 = wmma_bf16(qf[kc], kb1, s1);
        }

        // ---- mask + online softmax (rows live in 16-lane column groups) ----
        const bool k0 = (mrow[j0 + cl]      != 0);
        const bool k1 = (mrow[j0 + 16 + cl] != 0);
        float corr[8];
#pragma unroll
        for (int r = 0; r < 8; ++r) {
            float a0 = k0 ? s0[r] * scl : -3.0e38f;
            float a1 = k1 ? s1[r] * scl : -3.0e38f;
            float rm = fmaxf(a0, a1);
#pragma unroll
            for (int off = 1; off < 16; off <<= 1)
                rm = fmaxf(rm, __shfl_xor(rm, off, 32));
            const float mn  = fmaxf(m_r[r], rm);
            const float co  = __expf(m_r[r] - mn);
            const float p0e = __expf(a0 - mn);
            const float p1e = __expf(a1 - mn);
            float rs = p0e + p1e;
#pragma unroll
            for (int off = 1; off < 16; off <<= 1)
                rs += __shfl_xor(rs, off, 32);
            l_r[r]  = l_r[r] * co + rs;
            m_r[r]  = mn;
            corr[r] = co;
            const int prow = r + 8 * half;
            ldsP[wid][prow * 32 + cl]      = (__bf16)p0e;
            ldsP[wid][prow * 32 + 16 + cl] = (__bf16)p1e;
        }

        // rescale output accumulators by exp(m_old - m_new)
#pragma unroll
        for (int t = 0; t < 16; ++t)
#pragma unroll
            for (int r = 0; r < 8; ++r) o[t][r] *= corr[r];

        // ---- P (16x32 bf16) as A-fragment from LDS (C->A layout swap) ----
        v16bf pf;
        {
            const __bf16* pp = &ldsP[wid][cl * 32 + 16 * half];
#pragma unroll
            for (int i = 0; i < 16; ++i) { pf[i] = pp[i]; }
        }

        // ---- O += P . V_tile (V^T tile in LDS -> contiguous B-fragments) ----
#pragma unroll
        for (int t = 0; t < 16; ++t) {
            v16bf vb;
            const __bf16* pv = Vl + (size_t)(t * 16 + cl) * 32 + 16 * half;
#pragma unroll
            for (int i = 0; i < 16; ++i) vb[i] = pv[i];
            o[t] = wmma_bf16(pf, vb, o[t]);
        }

        __syncthreads();   // all waves done with tile `cur` before it is re-staged
    }

    // ---- epilogue: divide by softmax denominator, write f32 ----
    float* orow = out + ((size_t)(b * NA + m0)) * HID;
#pragma unroll
    for (int r = 0; r < 8; ++r) {
        const float inv = 1.0f / l_r[r];
        const int   row = r + 8 * half;
#pragma unroll
        for (int t = 0; t < 16; ++t)
            orow[(size_t)row * HID + t * 16 + cl] = o[t][r] * inv;
    }
}

// ---------------------------------------------------------------------------
extern "C" void kernel_launch(void* const* d_in, const int* in_sizes, int n_in,
                              void* d_out, int out_size, void* d_ws, size_t ws_size,
                              hipStream_t stream) {
    (void)in_sizes; (void)n_in; (void)out_size; (void)ws_size;
    const float* A   = (const float*)d_in[0];
    const float* Bx  = (const float*)d_in[1];
    const int*   msk = (const int*)d_in[2];
    const float* Wq  = (const float*)d_in[3];
    const float* bq  = (const float*)d_in[4];
    const float* Wk  = (const float*)d_in[5];
    const float* bk  = (const float*)d_in[6];
    const float* Wv  = (const float*)d_in[7];
    const float* bv  = (const float*)d_in[8];
    float* out = (float*)d_out;

    // workspace: Q (8MB) + K (8MB) + V^T (8MB) bf16
    __bf16* Qw = (__bf16*)d_ws;
    __bf16* Kw = Qw + (size_t)MROWS * HID;
    __bf16* Vt = Kw + (size_t)KROWS * HID;

    const int tiles = 3 * (MROWS / 16) * (HID / 16);      // 49152 waves
    qkv_proj<<<tiles / 8, 256, 0, stream>>>(A, Bx, Wq, bq, Wk, bk, Wv, bv, Qw, Kw, Vt);
    flash_attn<<<(MROWS / 16) / 8, 256, 0, stream>>>(Qw, Kw, Vt, msk, out);
}
